// RNNDecoder_91027536871753
// MI455X (gfx1250) — matrix operand order repaired and checked by reading, not compiled, not run on hardware
//
#include <hip/hip_runtime.h>
#include <math.h>

// ---------------------------------------------------------------------------
// Types for CDNA5 WMMA (wave32): v_wmma_f32_16x16x32_bf16
// A-frag: lane<16 -> row M=lane, K chunks [k+0..8)+[k+16..24); lane>=16 -> +8
// B-frag: lane<16 -> col N=lane, K [k+0..8)+[k+16..24); lane>=16 -> +8
// D-frag: vgpr r -> row r (+8 for lanes 16..31), col = lane&15
// ---------------------------------------------------------------------------
typedef __attribute__((ext_vector_type(16))) __bf16 bf16x16;
typedef __attribute__((ext_vector_type(8)))  float  f32x8;
typedef __attribute__((ext_vector_type(4)))  int    i32x4;

union FragU { bf16x16 v; i32x4 q[2]; };

__device__ inline unsigned short f2bf(float f) {
    unsigned int u = __float_as_uint(f);
    u += 0x7fffu + ((u >> 16) & 1u);      // round-to-nearest-even
    return (unsigned short)(u >> 16);
}

__device__ inline float sigf(float x) { return 1.0f / (1.0f + __expf(-x)); }

// Raw LDS byte offset of a __shared__ object (as(3) ptr value == LDS offset).
typedef __attribute__((address_space(3))) const char lds_cchar;
__device__ inline unsigned lds_off(const void* p) {
    return (unsigned)(unsigned long long)(lds_cchar*)p;
}

// ---------------------------------------------------------------------------
// GEMM: OUT[M,N] = A0[M,K]*B0[N,K]^T (+ A1[M,K]*B1[N,K]^T) + bias[N]
// A row-major bf16 (direct global b128 loads), B row-major bf16 [N,K] staged
// through LDS via async global->LDS copies (ASYNCcnt), double-buffered.
// MODE 0: f32 store outf[M,N]; MODE 1: bf16 store outb[M,N];
// MODE 2: f32 transpose-store row=(l*Bm+b) -> outf[b*Lm*Cm + l*Cm + col], col<Cm
// Block = 128 threads = 4 waves; wave tile 32(M)x64(N); block tile 128x64.
// K must be a multiple of 128.
// ---------------------------------------------------------------------------
#define KC        128                 // K-chunk staged per LDS buffer
#define COLSTRIDE 272                 // KC*2 bytes + 16B pad (bank-conflict-free)
#define BUFSZ     (64 * COLSTRIDE)    // 17408 B per buffer

template <int MODE>
__global__ __launch_bounds__(128) void k_gemm(
    const unsigned short* __restrict__ A0, const unsigned short* __restrict__ B0,
    const unsigned short* __restrict__ A1, const unsigned short* __restrict__ B1,
    const float* __restrict__ bias,
    float* __restrict__ outf, unsigned short* __restrict__ outb,
    int M, int N, int K, int Cm, int Lm, int Bm)
{
    __shared__ char smem[2 * BUFSZ];

    const int tid  = threadIdx.x;
    const int lane = tid & 31;
    const int wave = tid >> 5;
    const int half = lane >> 4;
    const int l16  = lane & 15;
    const int nBase = blockIdx.x * 64;
    const int mBase = blockIdx.y * 128 + wave * 32;
    const unsigned smBase = lds_off(smem);

    // async-copy: 64 cols x KC ushorts of B-chunk -> LDS buffer `buf`
    auto issueB = [&](const unsigned short* Bp, int kc, int buf) {
        const unsigned bb = smBase + (buf ? BUFSZ : 0);
#pragma unroll
        for (int r = 0; r < 8; ++r) {
            const int i   = tid + r * 128;   // 0..1023
            const int col = i >> 4;          // 64 cols
            const int seg = i & 15;          // 16 x 8-ushort segments
            const unsigned short* g = Bp + (size_t)(nBase + col) * K + kc + seg * 8;
            const unsigned lo = bb + col * COLSTRIDE + seg * 16;
            asm volatile("global_load_async_to_lds_b128 %0, %1, off"
                         :: "v"(lo), "v"(g) : "memory");
        }
    };

    f32x8 acc[2][4] = {};

    for (int pass = 0; pass < 2; ++pass) {
        const unsigned short* Ap = pass ? A1 : A0;
        const unsigned short* Bp = pass ? B1 : B0;
        if (Ap == nullptr) break;
        const unsigned short* arow0 = Ap + (size_t)(mBase + l16) * K;
        const unsigned short* arow1 = arow0 + (size_t)16 * K;
        const int NCH = K / KC;

        issueB(Bp, 0, 0);
        asm volatile("s_wait_asynccnt 0x0" ::: "memory");
        __syncthreads();

        for (int c = 0; c < NCH; ++c) {
            if (c + 1 < NCH) issueB(Bp, (c + 1) * KC, (c + 1) & 1);
            const char* bbuf = smem + ((c & 1) ? BUFSZ : 0);
#pragma unroll
            for (int kk = 0; kk < KC; kk += 32) {
                const int k = c * KC + kk;
                // global A loads first (longest latency)
                FragU fa0, fa1;
                const unsigned short* pa0 = arow0 + k + half * 8;
                fa0.q[0] = *(const i32x4*)(pa0);
                fa0.q[1] = *(const i32x4*)(pa0 + 16);
                const unsigned short* pa1 = arow1 + k + half * 8;
                fa1.q[0] = *(const i32x4*)(pa1);
                fa1.q[1] = *(const i32x4*)(pa1 + 16);
                // all 4 B-fragments into DISTINCT registers so the ds_load_b128s
                // can issue as one clause and retire under partial s_wait_dscnt
                FragU fb[4];
                const char* bkk = bbuf + l16 * COLSTRIDE + (kk + half * 8) * 2;
#pragma unroll
                for (int nt = 0; nt < 4; ++nt) {
                    const i32x4* pb = (const i32x4*)(bkk + nt * 16 * COLSTRIDE);
                    fb[nt].q[0] = pb[0];
                    fb[nt].q[1] = pb[2];
                }
#pragma unroll
                for (int nt = 0; nt < 4; ++nt) {
                    acc[0][nt] = __builtin_amdgcn_wmma_f32_16x16x32_bf16(
                        false, fa0.v, false, fb[nt].v, (short)0, acc[0][nt], false, false);
                    acc[1][nt] = __builtin_amdgcn_wmma_f32_16x16x32_bf16(
                        false, fa1.v, false, fb[nt].v, (short)0, acc[1][nt], false, false);
                }
            }
            asm volatile("s_wait_asynccnt 0x0" ::: "memory");
            __syncthreads();
        }
    }

#pragma unroll
    for (int mt = 0; mt < 2; ++mt) {
#pragma unroll
        for (int nt = 0; nt < 4; ++nt) {
            const int col  = nBase + nt * 16 + l16;
            const int rowb = mBase + mt * 16 + half * 8;
            if (MODE == 2 && col >= Cm) continue;
            const float bv = bias ? bias[col] : 0.0f;
#pragma unroll
            for (int r = 0; r < 8; ++r) {
                const float v = acc[mt][nt][r] + bv;
                const int row = rowb + r;
                if (MODE == 0) {
                    outf[(size_t)row * N + col] = v;
                } else if (MODE == 1) {
                    outb[(size_t)row * N + col] = f2bf(v);
                } else {
                    const int l = row / Bm;
                    const int b = row - l * Bm;
                    outf[((size_t)b * Lm + l) * Cm + col] = v;
                }
            }
        }
    }
}

// ---------------------------------------------------------------------------
// Helpers
// ---------------------------------------------------------------------------
__global__ void k_cvt(const float* __restrict__ s, unsigned short* __restrict__ d, int n) {
    int i = blockIdx.x * blockDim.x + threadIdx.x;
    if (i < n) d[i] = f2bf(s[i]);
}

__global__ void k_cvt_pad(const float* __restrict__ s, unsigned short* __restrict__ d,
                          int validRows, int K, int totRows) {
    int i = blockIdx.x * blockDim.x + threadIdx.x;
    int n = totRows * K;
    if (i >= n) return;
    int row = i / K;
    d[i] = (row < validRows) ? f2bf(s[i]) : (unsigned short)0;
}

__global__ void k_vecadd(const float* __restrict__ a, const float* __restrict__ b,
                         float* __restrict__ o, int n) {
    int i = blockIdx.x * blockDim.x + threadIdx.x;
    if (i < n) o[i] = a[i] + b[i];
}

// a0[row] = dot(W[row,:], x) + bsum[row]   (one wave per row)
__global__ void k_gemv_a0(const float* __restrict__ W, const float* __restrict__ x,
                          const float* __restrict__ bsum, float* __restrict__ out,
                          int rows, int K) {
    int row  = blockIdx.x * (blockDim.x >> 5) + (threadIdx.x >> 5);
    int lane = threadIdx.x & 31;
    if (row >= rows) return;
    const float* wr = W + (size_t)row * K;
    float s = 0.0f;
    for (int k = lane; k < K; k += 32) s += wr[k] * x[k];
    for (int off = 16; off > 0; off >>= 1) s += __shfl_xor(s, off, 32);
    if (lane == 0) out[row] = s + bsum[row];
}

// LSTM cell elementwise: gates [B,4H] (i,f,g,o), c_prev nullable (=> 0)
__global__ void k_cell(const float* __restrict__ gates, const float* __restrict__ c_prev,
                       float* __restrict__ c_out, unsigned short* __restrict__ h_out,
                       int Bn, int Hn) {
    int idx = blockIdx.x * blockDim.x + threadIdx.x;
    if (idx >= Bn * Hn) return;
    int b = idx / Hn, j = idx - b * Hn;
    const float* g = gates + (size_t)b * 4 * Hn;
    float gi = g[j], gf = g[Hn + j], gg = g[2 * Hn + j], go = g[3 * Hn + j];
    float c  = c_prev ? c_prev[idx] : 0.0f;
    float cn = sigf(gf) * c + sigf(gi) * tanhf(gg);
    c_out[idx] = cn;
    h_out[idx] = f2bf(sigf(go) * tanhf(cn));
}

// ---------------------------------------------------------------------------
// Launcher
// ---------------------------------------------------------------------------
extern "C" void kernel_launch(void* const* d_in, const int* in_sizes, int n_in,
                              void* d_out, int out_size, void* d_ws, size_t ws_size,
                              hipStream_t stream) {
    (void)in_sizes; (void)n_in; (void)out_size; (void)ws_size;
    const int Bb = 128, IN = 512, H = 1024, G = 4096, Cc = 10000, Np = 10048, Ll = 50;

    const float* enc_in_f = (const float*)d_in[0];
    const float* W_enc = (const float*)d_in[1];
    const float* b_enc = (const float*)d_in[2];
    const float* W_ih0 = (const float*)d_in[3];
    const float* W_hh0 = (const float*)d_in[4];
    const float* b_ih0 = (const float*)d_in[5];
    const float* b_hh0 = (const float*)d_in[6];
    const float* W_ih1 = (const float*)d_in[7];
    const float* W_hh1 = (const float*)d_in[8];
    const float* b_ih1 = (const float*)d_in[9];
    const float* b_hh1 = (const float*)d_in[10];
    const float* W_out = (const float*)d_in[11];
    const float* b_out = (const float*)d_in[12];
    float* out = (float*)d_out;

    char* ws = (char*)d_ws;
    size_t off = 0;
    auto alloc = [&](size_t bytes) -> char* {
        char* p = ws + off;
        off += (bytes + 255) & ~(size_t)255;
        return p;
    };
    unsigned short* wenc_bf  = (unsigned short*)alloc((size_t)H * IN * 2);
    unsigned short* wih0_bf  = (unsigned short*)alloc((size_t)G * H * 2);
    unsigned short* whh0_bf  = (unsigned short*)alloc((size_t)G * H * 2);
    unsigned short* wih1_bf  = (unsigned short*)alloc((size_t)G * H * 2);
    unsigned short* whh1_bf  = (unsigned short*)alloc((size_t)G * H * 2);
    unsigned short* wout_bf  = (unsigned short*)alloc((size_t)Np * H * 2);
    unsigned short* encin_bf = (unsigned short*)alloc((size_t)Bb * IN * 2);
    unsigned short* enc_bf   = (unsigned short*)alloc((size_t)Bb * H * 2);
    unsigned short* h0_bf    = (unsigned short*)alloc((size_t)Bb * H * 2);
    unsigned short* hs_bf    = (unsigned short*)alloc((size_t)Ll * Bb * H * 2);
    float* a0     = (float*)alloc((size_t)G * 4);
    float* b0sum  = (float*)alloc((size_t)G * 4);
    float* b1sum  = (float*)alloc((size_t)G * 4);
    float* gates0 = (float*)alloc((size_t)Bb * G * 4);
    float* gates1 = (float*)alloc((size_t)Bb * G * 4);
    float* c0     = (float*)alloc((size_t)Bb * H * 4);
    float* c1     = (float*)alloc((size_t)Bb * H * 4);

    auto cgrid = [](int n) { return (n + 255) / 256; };

    // Precompute (re-done every call for determinism)
    k_cvt<<<cgrid(H * IN), 256, 0, stream>>>(W_enc, wenc_bf, H * IN);
    k_cvt<<<cgrid(G * H), 256, 0, stream>>>(W_ih0, wih0_bf, G * H);
    k_cvt<<<cgrid(G * H), 256, 0, stream>>>(W_hh0, whh0_bf, G * H);
    k_cvt<<<cgrid(G * H), 256, 0, stream>>>(W_ih1, wih1_bf, G * H);
    k_cvt<<<cgrid(G * H), 256, 0, stream>>>(W_hh1, whh1_bf, G * H);
    k_cvt_pad<<<cgrid(Np * H), 256, 0, stream>>>(W_out, wout_bf, Cc, H, Np);
    k_cvt<<<cgrid(Bb * IN), 256, 0, stream>>>(enc_in_f, encin_bf, Bb * IN);
    k_vecadd<<<cgrid(G), 256, 0, stream>>>(b_ih0, b_hh0, b0sum, G);
    k_vecadd<<<cgrid(G), 256, 0, stream>>>(b_ih1, b_hh1, b1sum, G);
    // a0 = W_ih0 @ b_enc + b_ih0 + b_hh0  (constant layer-0 input term, t>=1)
    k_gemv_a0<<<G / 8, 256, 0, stream>>>(W_ih0, b_enc, b0sum, a0, G, H);

    dim3 blk(128);
    // enc_proj (bf16) = encoder_output @ W_enc^T + b_enc   [128,1024], K=512
    k_gemm<1><<<dim3(H / 64, Bb / 128), blk, 0, stream>>>(
        encin_bf, wenc_bf, nullptr, nullptr, b_enc, nullptr, enc_bf,
        Bb, H, IN, 0, 0, 0);

    for (int t = 0; t < Ll; ++t) {
        // layer 0 gates
        if (t == 0) {
            // x = enc_proj, h0 = enc_proj  -> dual GEMM, same A
            k_gemm<0><<<dim3(G / 64, 1), blk, 0, stream>>>(
                enc_bf, wih0_bf, enc_bf, whh0_bf, b0sum, gates0, nullptr,
                Bb, G, H, 0, 0, 0);
        } else {
            // x = b_enc (constant) folded into a0 -> single recurrent GEMM
            k_gemm<0><<<dim3(G / 64, 1), blk, 0, stream>>>(
                h0_bf, whh0_bf, nullptr, nullptr, a0, gates0, nullptr,
                Bb, G, H, 0, 0, 0);
        }
        k_cell<<<cgrid(Bb * H), 256, 0, stream>>>(
            gates0, (t == 0) ? nullptr : c0, c0, h0_bf, Bb, H);

        // layer 1 gates: h0_new @ W_ih1^T + h1_prev @ W_hh1^T + bias
        const unsigned short* h1prev =
            (t == 0) ? enc_bf : (hs_bf + (size_t)(t - 1) * Bb * H);
        k_gemm<0><<<dim3(G / 64, 1), blk, 0, stream>>>(
            h0_bf, wih1_bf, h1prev, whh1_bf, b1sum, gates1, nullptr,
            Bb, G, H, 0, 0, 0);
        k_cell<<<cgrid(Bb * H), 256, 0, stream>>>(
            gates1, (t == 0) ? nullptr : c1, c1, hs_bf + (size_t)t * Bb * H, Bb, H);
    }

    // logits: hs [6400,1024] @ W_out^T (+b_out), stored transposed to [B,L,C]
    k_gemm<2><<<dim3(Np / 64, (Ll * Bb) / 128), blk, 0, stream>>>(
        hs_bf, wout_bf, nullptr, nullptr, b_out, out, nullptr,
        Ll * Bb, Np, H, Cc, Ll, Bb);
}
